// ConformerMHSAQuant_73924977099287
// MI455X (gfx1250) — matrix operand order repaired and checked by reading, not compiled, not run on hardware
//
#include <hip/hip_runtime.h>
#include <hip/hip_bf16.h>
#include <stdint.h>

typedef _Float16 f16;
typedef _Float16 h4  __attribute__((ext_vector_type(4)));
typedef _Float16 h8  __attribute__((ext_vector_type(8)));
typedef _Float16 h16 __attribute__((ext_vector_type(16)));
typedef float    f8  __attribute__((ext_vector_type(8)));

#define B_    16
#define T_    1024
#define F_    512
#define H_    8
#define DH    64
#define NTOK  (B_ * T_)        // 16384
#define NELEM (NTOK * F_)      // 8388608
#define G3F   (3 * F_)         // 1536

// ---------------- order-preserving float<->uint for atomic min/max ----------------
__device__ __forceinline__ unsigned enc_f(float f) {
  unsigned u = __float_as_uint(f);
  return (u & 0x80000000u) ? ~u : (u | 0x80000000u);
}
__device__ __forceinline__ float dec_f(unsigned u) {
  unsigned b = (u & 0x80000000u) ? (u & 0x7fffffffu) : ~u;
  return __uint_as_float(b);
}

// ---------------- per-tensor asymmetric fake-quant (forward value) ----------------
__device__ __forceinline__ float fq(float x, float xmin, float xmax) {
  float scale = (xmax - xmin) * (1.0f / 255.0f) + 1e-8f;
  float zp = rintf(-xmin / scale);
  float q  = fminf(fmaxf(rintf(x / scale) + zp, 0.0f), 255.0f);
  return (q - zp) * scale;
}

// ---------------- wave32 reductions ----------------
__device__ __forceinline__ float wsum32(float v) {
#pragma unroll
  for (int m = 16; m >= 1; m >>= 1) v += __shfl_xor(v, m, 32);
  return v;
}
__device__ __forceinline__ float wmin32(float v) {
#pragma unroll
  for (int m = 16; m >= 1; m >>= 1) v = fminf(v, __shfl_xor(v, m, 32));
  return v;
}
__device__ __forceinline__ float wmax32(float v) {
#pragma unroll
  for (int m = 16; m >= 1; m >>= 1) v = fmaxf(v, __shfl_xor(v, m, 32));
  return v;
}
// max across each 16-lane half (C-fragment rows live per half)
__device__ __forceinline__ float hmax16(float v) {
#pragma unroll
  for (int m = 8; m >= 1; m >>= 1) v = fmaxf(v, __shfl_xor(v, m, 32));
  return v;
}

// ---------------- WMMA fragment loaders (contiguous k order, same for A and B) ----------------
// A (16x32 f16, MxK): lane holds row=lane&15, K = c0..c0+15 contiguous (c0 = 0 / 16)
__device__ __forceinline__ h16 load_a_frag(const f16* base, int ld, int lane, int koff) {
  int r  = lane & 15;
  int c0 = (lane < 16) ? 0 : 16;
  const f16* p = base + r * ld + koff + c0;
  h8 lo = *reinterpret_cast<const h8*>(p);
  h8 hi = *reinterpret_cast<const h8*>(p + 8);
  h16 out;
#pragma unroll
  for (int i = 0; i < 8; i++) { out[i] = lo[i]; out[i + 8] = hi[i]; }
  return out;
}
// B (32x16 f16, KxN): lane holds N=lane&15, K = k0..k0+15 contiguous (k0 = 0 / 16)
__device__ __forceinline__ h16 load_b_frag(const f16* base, int ld, int lane, int koff) {
  int n  = lane & 15;
  int k0 = (lane < 16) ? 0 : 16;
  return *reinterpret_cast<const h16*>(base + n * ld + koff + k0);
}

// =====================================================================
// Kernel 0: init min/max slots (fake_quant clamps range to include 0)
// =====================================================================
__global__ void k_init_slots(unsigned* slots) {
  if (threadIdx.x < 4) slots[threadIdx.x] = enc_f(0.0f);
}

// =====================================================================
// Kernel 1: global min/max of x (float4 vectorized)
// =====================================================================
__global__ __launch_bounds__(256) void k_minmax(const float4* __restrict__ x4, int n4,
                                                unsigned* __restrict__ slots) {
  float lmin = 0.0f, lmax = 0.0f;  // final range clamped to include 0 anyway
  for (int i = blockIdx.x * blockDim.x + threadIdx.x; i < n4; i += gridDim.x * blockDim.x) {
    float4 v = x4[i];
    lmin = fminf(lmin, fminf(fminf(v.x, v.y), fminf(v.z, v.w)));
    lmax = fmaxf(lmax, fmaxf(fmaxf(v.x, v.y), fmaxf(v.z, v.w)));
  }
  lmin = wmin32(lmin);
  lmax = wmax32(lmax);
  if ((threadIdx.x & 31) == 0) {
    atomicMin(&slots[0], enc_f(lmin));
    atomicMax(&slots[1], enc_f(lmax));
  }
}

// =====================================================================
// Kernel 2: per-row stats of fake_quant(x) + L1-mean-center norm:
// store (mean, 1/denom) per row, accumulate global y min/max.
// One wave per row (F=512 = 128 float4). No f32 y is materialized.
// =====================================================================
__global__ __launch_bounds__(256) void k_stats(const float4* __restrict__ x4,
                                               const float4* __restrict__ g4,
                                               const float4* __restrict__ bb4,
                                               unsigned* __restrict__ slots,
                                               float2* __restrict__ stats) {
  int wave = threadIdx.x >> 5, lane = threadIdx.x & 31;
  int row = blockIdx.x * 8 + wave;
  float xmin = dec_f(slots[0]);
  float xmax = dec_f(slots[1]);
  const float4* xr = x4 + (size_t)row * (F_ / 4);
  float v[16];
#pragma unroll
  for (int j = 0; j < 4; j++) {
    float4 t = xr[lane + 32 * j];
    v[4 * j + 0] = fq(t.x, xmin, xmax);
    v[4 * j + 1] = fq(t.y, xmin, xmax);
    v[4 * j + 2] = fq(t.z, xmin, xmax);
    v[4 * j + 3] = fq(t.w, xmin, xmax);
  }
  float s = 0.0f;
#pragma unroll
  for (int j = 0; j < 16; j++) s += v[j];
  float mean = wsum32(s) * (1.0f / (float)F_);
  float a = 0.0f;
#pragma unroll
  for (int j = 0; j < 16; j++) { v[j] -= mean; a += fabsf(v[j]); }
  float inv = 1.0f / (wsum32(a) * (1.0f / (float)F_) + 1e-5f);
  float lmin = 0.0f, lmax = 0.0f;
#pragma unroll
  for (int j = 0; j < 4; j++) {
    float4 gg = g4[lane + 32 * j];
    float4 bv = bb4[lane + 32 * j];
    float o0 = v[4 * j + 0] * inv * gg.x + bv.x;
    float o1 = v[4 * j + 1] * inv * gg.y + bv.y;
    float o2 = v[4 * j + 2] * inv * gg.z + bv.z;
    float o3 = v[4 * j + 3] * inv * gg.w + bv.w;
    lmin = fminf(lmin, fminf(fminf(o0, o1), fminf(o2, o3)));
    lmax = fmaxf(lmax, fmaxf(fmaxf(o0, o1), fmaxf(o2, o3)));
  }
  lmin = wmin32(lmin);
  lmax = wmax32(lmax);
  if (lane == 0) {
    stats[row] = make_float2(mean, inv);
    atomicMin(&slots[2], enc_f(lmin));
    atomicMax(&slots[3], enc_f(lmax));
  }
}

// =====================================================================
// Kernel 3: recompute y from x + per-row stats, fake_quant(y) -> f16.
// One wave per row; x is L2-resident from the previous pass.
// =====================================================================
__global__ __launch_bounds__(256) void k_quant_row(const float4* __restrict__ x4,
                                                   const float4* __restrict__ g4,
                                                   const float4* __restrict__ bb4,
                                                   const unsigned* __restrict__ slots,
                                                   const float2* __restrict__ stats,
                                                   f16* __restrict__ yq) {
  int wave = threadIdx.x >> 5, lane = threadIdx.x & 31;
  int row = blockIdx.x * 8 + wave;
  float xmin = dec_f(slots[0]);
  float xmax = dec_f(slots[1]);
  float ymin = dec_f(slots[2]);
  float ymax = dec_f(slots[3]);
  float2 st = stats[row];
  const float4* xr = x4 + (size_t)row * (F_ / 4);
  f16* yr = yq + (size_t)row * F_;
#pragma unroll
  for (int j = 0; j < 4; j++) {
    float4 t = xr[lane + 32 * j];
    float4 gg = g4[lane + 32 * j];
    float4 bv = bb4[lane + 32 * j];
    h4 o;
    o[0] = (f16)fq((fq(t.x, xmin, xmax) - st.x) * st.y * gg.x + bv.x, ymin, ymax);
    o[1] = (f16)fq((fq(t.y, xmin, xmax) - st.x) * st.y * gg.y + bv.y, ymin, ymax);
    o[2] = (f16)fq((fq(t.z, xmin, xmax) - st.x) * st.y * gg.z + bv.z, ymin, ymax);
    o[3] = (f16)fq((fq(t.w, xmin, xmax) - st.x) * st.y * gg.w + bv.w, ymin, ymax);
    *reinterpret_cast<h4*>(&yr[(lane + 32 * j) * 4]) = o;
  }
}

// f32 -> f16 convert (weights), 8 elements/thread
__global__ __launch_bounds__(256) void k_cvt(const float4* __restrict__ a4,
                                             h8* __restrict__ o8, int n8) {
  for (int i = blockIdx.x * blockDim.x + threadIdx.x; i < n8; i += gridDim.x * blockDim.x) {
    float4 a = a4[2 * i], b = a4[2 * i + 1];
    h8 o;
    o[0] = (f16)a.x; o[1] = (f16)a.y; o[2] = (f16)a.z; o[3] = (f16)a.w;
    o[4] = (f16)b.x; o[5] = (f16)b.y; o[6] = (f16)b.z; o[7] = (f16)b.w;
    o8[i] = o;
  }
}

// =====================================================================
// WMMA GEMM: C[M,N] = A[M,K] * W[N,K]^T + bias
// 256 thr (8 waves, 2x4 grid), block tile 128x128, BK=64, double-buffered
// LDS, wave tile 64x32 (4x2 frags) -> 16 WMMAs per wave per K-step.
// MODE 0: f32 out[M,N].  MODE 1: scatter QKV; V stored transposed [B,H,Dh,T].
// =====================================================================
__device__ __forceinline__ void stage_tiles(const f16* __restrict__ A,
                                            const f16* __restrict__ W,
                                            f16* sA, f16* sB,
                                            int bm, int bn, int K, int k0, int tid) {
  int row = tid >> 1;
  int cs = (tid & 1) * 32;  // halfs
  const float4* sa = reinterpret_cast<const float4*>(A + (size_t)(bm + row) * K + k0 + cs);
  float4* da = reinterpret_cast<float4*>(&sA[row * 64 + cs]);
  da[0] = sa[0]; da[1] = sa[1]; da[2] = sa[2]; da[3] = sa[3];
  const float4* sb = reinterpret_cast<const float4*>(W + (size_t)(bn + row) * K + k0 + cs);
  float4* db = reinterpret_cast<float4*>(&sB[row * 64 + cs]);
  db[0] = sb[0]; db[1] = sb[1]; db[2] = sb[2]; db[3] = sb[3];
}

template <int MODE>
__global__ __launch_bounds__(256) void k_gemm(const f16* __restrict__ A,
                                              const f16* __restrict__ W,
                                              const float* __restrict__ bias,
                                              float* __restrict__ out,
                                              f16* __restrict__ qf, f16* __restrict__ kf,
                                              f16* __restrict__ vt,
                                              int M, int N, int K) {
  __shared__ alignas(32) f16 lA[2][128 * 64];
  __shared__ alignas(32) f16 lB[2][128 * 64];
  int tid = threadIdx.x, lane = tid & 31, w = tid >> 5;
  int wm = w & 1, wn = w >> 1;  // 2 x 4 wave grid
  int bm = blockIdx.x * 128, bn = blockIdx.y * 128;
  f8 acc[4][2] = {};
  int nk = K >> 6;

  stage_tiles(A, W, lA[0], lB[0], bm, bn, K, 0, tid);
  __syncthreads();

  for (int kt = 0; kt < nk; kt++) {
    int cur = kt & 1;
    if (kt + 1 < nk)  // overlap next tile's global loads with this tile's WMMAs
      stage_tiles(A, W, lA[cur ^ 1], lB[cur ^ 1], bm, bn, K, (kt + 1) << 6, tid);
    const f16* cA = lA[cur];
    const f16* cB = lB[cur];
#pragma unroll
    for (int kk = 0; kk < 2; kk++) {
      h16 af[4], bf[2];
#pragma unroll
      for (int mf = 0; mf < 4; mf++)
        af[mf] = load_a_frag(cA + (wm * 64 + mf * 16) * 64, 64, lane, kk * 32);
#pragma unroll
      for (int nf = 0; nf < 2; nf++)
        bf[nf] = load_b_frag(cB + (wn * 32 + nf * 16) * 64, 64, lane, kk * 32);
#pragma unroll
      for (int mf = 0; mf < 4; mf++)
#pragma unroll
        for (int nf = 0; nf < 2; nf++)
          acc[mf][nf] = __builtin_amdgcn_wmma_f32_16x16x32_f16(
              false, af[mf], false, bf[nf], (short)0, acc[mf][nf], false, false);
    }
    __syncthreads();
  }

  int hb = (lane < 16) ? 0 : 8;
  int col = lane & 15;
#pragma unroll
  for (int mf = 0; mf < 4; mf++) {
#pragma unroll
    for (int nf = 0; nf < 2; nf++) {
      int ncol = bn + wn * 32 + nf * 16 + col;
      float bv = bias[ncol];
#pragma unroll
      for (int r = 0; r < 8; r++) {
        int mrow = bm + wm * 64 + mf * 16 + r + hb;
        float val = acc[mf][nf][r] + bv;
        if (MODE == 0) {
          out[(size_t)mrow * N + ncol] = val;
        } else {
          int which = ncol >> 9;  // 0=q,1=k,2=v
          int f = ncol & 511;
          int hh = f >> 6;
          int d = f & 63;
          int b = mrow >> 10;
          int t = mrow & 1023;
          f16 hv = (f16)val;
          if (which == 0)
            qf[(((size_t)b * H_ + hh) * T_ + t) * DH + d] = hv;
          else if (which == 1)
            kf[(((size_t)b * H_ + hh) * T_ + t) * DH + d] = hv;
          else  // V transposed: [B,H,Dh,T] for direct B-fragment loads in attention
            vt[(((size_t)b * H_ + hh) * DH + d) * T_ + t] = hv;
        }
      }
    }
  }
}

// =====================================================================
// Flash attention. grid=(T/128, B*H), 256 thr (8 waves), NO __syncthreads:
// K/V fragments load straight from L2-resident global memory, the only LDS
// is the wave-private P tile (same-wave DS ordering + explicit s_wait_dscnt).
// Row-sums come from a WMMA against an all-ones B matrix (no shuffle sums).
// =====================================================================
__global__ __launch_bounds__(256) void k_attn(const f16* __restrict__ qf,
                                              const f16* __restrict__ kf,
                                              const f16* __restrict__ vt,
                                              const unsigned char* __restrict__ mask,
                                              f16* __restrict__ ctxf) {
  __shared__ alignas(32) f16 lP[8 * 16 * 64];  // per-wave P tile [m][key], 16KB
  int tid = threadIdx.x, lane = tid & 31, w = tid >> 5;
  int bh = blockIdx.y;
  int b = bh >> 3;
  int h = bh & 7;
  int q0 = blockIdx.x * 128 + w * 16;
  size_t base = (size_t)bh * T_ * DH;  // same element count for qf/kf/vt

  h16 qa[2];
  qa[0] = load_a_frag(qf + base + (size_t)q0 * DH, DH, lane, 0);
  qa[1] = load_a_frag(qf + base + (size_t)q0 * DH, DH, lane, 32);

  h16 ones;
#pragma unroll
  for (int i = 0; i < 16; i++) ones[i] = (f16)1.0f;

  f8 ctx[4] = {};
  float mrow[8], lrow[8];
#pragma unroll
  for (int r = 0; r < 8; r++) { mrow[r] = -3.0e38f; lrow[r] = 0.0f; }
  const float sc = 0.125f;  // 1/sqrt(Dh)
  f16* pT = lP + w * 16 * 64;
  int hb = (lane < 16) ? 0 : 8;
  int col = lane & 15;

  for (int kb = 0; kb < T_; kb += 64) {
    // ---- scores: 4 N-frags of 16 keys, K reduced over Dh=64 (8 WMMAs) ----
    f8 s[4] = {};
#pragma unroll
    for (int nb = 0; nb < 4; nb++) {
      const f16* kbase = kf + base + (size_t)(kb + nb * 16) * DH;
#pragma unroll
      for (int kk = 0; kk < 2; kk++) {
        h16 kbf = load_b_frag(kbase, DH, lane, kk * 32);
        s[nb] = __builtin_amdgcn_wmma_f32_16x16x32_f16(false, qa[kk], false, kbf,
                                                       (short)0, s[nb], false, false);
      }
    }
    bool vld[4];
#pragma unroll
    for (int nb = 0; nb < 4; nb++) vld[nb] = mask[b * T_ + kb + nb * 16 + col] != 0;

    // ---- online softmax: shuffle only for row-max ----
#pragma unroll
    for (int r = 0; r < 8; r++) {
      float a[4];
#pragma unroll
      for (int nb = 0; nb < 4; nb++) a[nb] = vld[nb] ? s[nb][r] * sc : -1.0e9f;
      float rmax = hmax16(fmaxf(fmaxf(a[0], a[1]), fmaxf(a[2], a[3])));
      float mn = fmaxf(mrow[r], rmax);
      float corr = __expf(mrow[r] - mn);
      mrow[r] = mn;
      lrow[r] *= corr;  // row-sum added below via ones-WMMA
#pragma unroll
      for (int d = 0; d < 4; d++) ctx[d][r] *= corr;
#pragma unroll
      for (int nb = 0; nb < 4; nb++)
        pT[(r + hb) * 64 + nb * 16 + col] = (f16)__expf(a[nb] - mn);
    }
    asm volatile("s_wait_dscnt 0" ::: "memory");  // wave-local LDS RAW fence

    h16 pa0 = load_a_frag(pT, 64, lane, 0);
    h16 pa1 = load_a_frag(pT, 64, lane, 32);

    // ---- row-sum of P via WMMA with all-ones B (2 WMMAs, no shuffles) ----
    f8 rs = {};
    rs = __builtin_amdgcn_wmma_f32_16x16x32_f16(false, pa0, false, ones, (short)0, rs, false, false);
    rs = __builtin_amdgcn_wmma_f32_16x16x32_f16(false, pa1, false, ones, (short)0, rs, false, false);
#pragma unroll
    for (int r = 0; r < 8; r++) lrow[r] += rs[r];

    // ---- ctx += P x V using pre-transposed V [Dh][T] (8 WMMAs) ----
#pragma unroll
    for (int d = 0; d < 4; d++) {
      const f16* vbase = vt + base + (size_t)(d * 16) * T_;
      h16 vb0 = load_b_frag(vbase, T_, lane, kb);
      h16 vb1 = load_b_frag(vbase, T_, lane, kb + 32);
      ctx[d] = __builtin_amdgcn_wmma_f32_16x16x32_f16(false, pa0, false, vb0, (short)0, ctx[d], false, false);
      ctx[d] = __builtin_amdgcn_wmma_f32_16x16x32_f16(false, pa1, false, vb1, (short)0, ctx[d], false, false);
    }
  }

  // ---- normalize, write ctx as [B*T, F] f16 (F index = h*64 + d) ----
#pragma unroll
  for (int r = 0; r < 8; r++) {
    float invl = lrow[r] > 0.0f ? 1.0f / lrow[r] : 0.0f;
    int tok = b * T_ + q0 + r + hb;
#pragma unroll
    for (int d = 0; d < 4; d++)
      ctxf[(size_t)tok * F_ + h * DH + d * 16 + col] = (f16)(ctx[d][r] * invl);
  }
}

// =====================================================================
// Host launcher
// =====================================================================
extern "C" void kernel_launch(void* const* d_in, const int* in_sizes, int n_in,
                              void* d_out, int out_size, void* d_ws, size_t ws_size,
                              hipStream_t stream) {
  const float* x          = (const float*)d_in[0];
  const unsigned char* mk = (const unsigned char*)d_in[1];  // bool mask
  const float* g          = (const float*)d_in[2];
  const float* bb         = (const float*)d_in[3];
  const float* wqkv       = (const float*)d_in[4];
  const float* bqkv       = (const float*)d_in[5];
  const float* wout       = (const float*)d_in[6];
  const float* bout       = (const float*)d_in[7];
  float* out = (float*)d_out;

  char* ws = (char*)d_ws;
  size_t off = 0;
  auto alloc = [&](size_t bytes) -> char* {
    char* p = ws + off;
    off += (bytes + 255) & ~(size_t)255;
    return p;
  };
  unsigned* slots = (unsigned*)alloc(64);
  float2* stats = (float2*)alloc((size_t)NTOK * sizeof(float2));
  f16* yq   = (f16*)alloc((size_t)NELEM * 2);
  f16* wq16 = (f16*)alloc((size_t)G3F * F_ * 2);
  f16* wo16 = (f16*)alloc((size_t)F_ * F_ * 2);
  f16* qh   = (f16*)alloc((size_t)NELEM * 2);
  f16* kh   = (f16*)alloc((size_t)NELEM * 2);
  f16* vth  = (f16*)alloc((size_t)NELEM * 2);  // V transposed [B,H,Dh,T]
  f16* ch   = (f16*)alloc((size_t)NELEM * 2);
  (void)ws_size; (void)in_sizes; (void)n_in; (void)out_size;

  k_init_slots<<<1, 32, 0, stream>>>(slots);
  k_minmax<<<1024, 256, 0, stream>>>((const float4*)x, NELEM / 4, slots);
  k_stats<<<NTOK / 8, 256, 0, stream>>>((const float4*)x, (const float4*)g,
                                        (const float4*)bb, slots, stats);
  k_quant_row<<<NTOK / 8, 256, 0, stream>>>((const float4*)x, (const float4*)g,
                                            (const float4*)bb, slots, stats, yq);
  k_cvt<<<512, 256, 0, stream>>>((const float4*)wqkv, (h8*)wq16, (G3F * F_) / 8);
  k_cvt<<<256, 256, 0, stream>>>((const float4*)wout, (h8*)wo16, (F_ * F_) / 8);
  k_gemm<1><<<dim3(NTOK / 128, G3F / 128), 256, 0, stream>>>(
      yq, wq16, bqkv, nullptr, qh, kh, vth, NTOK, G3F, F_);
  k_attn<<<dim3(T_ / 128, B_ * H_), 256, 0, stream>>>(qh, kh, vth, mk, ch);
  k_gemm<0><<<dim3(NTOK / 128, F_ / 128), 256, 0, stream>>>(
      ch, wo16, bout, out, nullptr, nullptr, nullptr, NTOK, F_, F_);
}
